// RNN_86320252715682
// MI455X (gfx1250) — matrix operand order repaired and checked by reading, not compiled, not run on hardware
//
#include <hip/hip_runtime.h>
#include <hip/hip_bf16.h>

// ---------------------------------------------------------------------------
// tanh-RNN for MI455X (gfx1250, wave32, WMMA + async global->LDS staging)
//   xv = x @ V + b + b2            (32768 x 256 x 1024 GEMM)
//   h_t = tanh(h_{t-1} @ W + xv_t) (512 sequential 64 x 1024 x 1024 GEMMs)
// f32 split into bf16 hi + bf16 lo; 3 bf16 WMMAs per K-chunk (hi*hi + hi*lo +
// lo*hi) for ~f32 accuracy. Recurrence kernel is persistent (16 blocks) with
// an atomic grid barrier; operands are staged into LDS with
// GLOBAL_LOAD_ASYNC_TO_LDS_B128 double-buffering (ASYNCcnt pipeline).
// ---------------------------------------------------------------------------

typedef __attribute__((ext_vector_type(16))) __bf16 v16bf;
typedef __attribute__((ext_vector_type(8)))  __bf16 v8bf;
typedef __attribute__((ext_vector_type(8)))  float  v8f;
typedef __attribute__((ext_vector_type(4)))  float  v4f;
typedef __attribute__((ext_vector_type(4)))  int    v4i;

#define SEQ   512
#define BATCH 64
#define INSZ  256
#define HID   1024

// workspace layout (bytes)
#define OFF_CNT   0u
#define OFF_HHI   1024u                                  // 2 x 64x1024 bf16 (double buffer)
#define OFF_HLO   (OFF_HHI + 2u*BATCH*HID*2u)
#define OFF_WTHI  (OFF_HLO + 2u*BATCH*HID*2u)            // 1024x1024 bf16 (W transposed)
#define OFF_WTLO  (OFF_WTHI + (unsigned)(HID*HID*2))
#define OFF_VTHI  (OFF_WTLO + (unsigned)(HID*HID*2))     // 1024x256 bf16 (V transposed)
#define OFF_VTLO  (OFF_VTHI + (unsigned)(HID*INSZ*2))
// total ~5.8 MB

__device__ __forceinline__ v16bf cat16(v8bf a, v8bf b) {
  return __builtin_shufflevector(a, b, 0,1,2,3,4,5,6,7,8,9,10,11,12,13,14,15);
}

// async 16-byte global -> LDS copy (no VGPR data, tracked by ASYNCcnt)
__device__ __forceinline__ void async16(const __bf16* g, __bf16* l) {
  __builtin_amdgcn_global_load_async_to_lds_b128((v4i*)g, (v4i*)l,
                                                 /*ioffset=*/0, /*cpol=*/0);
}

// ---------------------------------------------------------------------------
// Prep: split/transpose W and V into bf16 hi/lo, zero h0 + barrier counter.
// Wt[n][k] = W[k][n] so the WMMA B-operand (16 contiguous K at fixed N) is a
// contiguous 32-byte per-lane load.
// ---------------------------------------------------------------------------
__global__ void rnn_prep(const float* __restrict__ W, const float* __restrict__ V,
                         unsigned* __restrict__ cnt,
                         __bf16* __restrict__ wthi, __bf16* __restrict__ wtlo,
                         __bf16* __restrict__ vthi, __bf16* __restrict__ vtlo,
                         __bf16* __restrict__ hhi,  __bf16* __restrict__ hlo) {
  const int idx    = blockIdx.x * blockDim.x + threadIdx.x;
  const int stride = gridDim.x * blockDim.x;
  if (idx == 0) *cnt = 0u;
  for (int i = idx; i < HID * HID; i += stride) {
    const int n = i / HID, k = i % HID;
    const float w = W[k * HID + n];
    const __bf16 h = (__bf16)w;
    wthi[i] = h;
    wtlo[i] = (__bf16)(w - (float)h);
  }
  for (int i = idx; i < HID * INSZ; i += stride) {
    const int n = i / INSZ, k = i % INSZ;
    const float v = V[k * HID + n];
    const __bf16 h = (__bf16)v;
    vthi[i] = h;
    vtlo[i] = (__bf16)(v - (float)h);
  }
  for (int i = idx; i < BATCH * HID; i += stride) {   // zero h0 (buffer 0)
    hhi[i] = (__bf16)0.0f;
    hlo[i] = (__bf16)0.0f;
  }
}

// ---------------------------------------------------------------------------
// xv = x @ V + b + b2, written into the h_seq region of d_out (overwritten in
// place by h_t later; each element is read exactly once by its owning lane).
// One wave per 16x16 output tile; 2048 m-tiles x 64 n-tiles. One-shot and
// bandwidth-trivial, so operands come straight from L2.
// ---------------------------------------------------------------------------
__global__ void __launch_bounds__(256) rnn_xv(
    const float* __restrict__ x,
    const __bf16* __restrict__ vthi, const __bf16* __restrict__ vtlo,
    const float* __restrict__ b, const float* __restrict__ b2,
    float* __restrict__ out) {
  const int lane = threadIdx.x & 31;
  const int wave = blockIdx.x * (blockDim.x >> 5) + (threadIdx.x >> 5);
  const int mt = wave >> 6;            // 0..2047 (row tile over 512*64 rows)
  const int nt = wave & 63;            // 0..63
  const int m0 = mt * 16, n0 = nt * 16;
  const int lm = lane & 15;
  const int lh = lane >> 4;            // 0/1: lane-half

  const float*  arow  = x    + (size_t)(m0 + lm) * INSZ;
  const __bf16* bhrow = vthi + (size_t)(n0 + lm) * INSZ;
  const __bf16* blrow = vtlo + (size_t)(n0 + lm) * INSZ;

  v8f acc = {};
  for (int kk = 0; kk < INSZ; kk += 32) {
    // A fragment: K = lh*16 + {0..15} within chunk
    const int ka = kk + lh * 16;
    float fa[16];
    *(v4f*)(fa + 0)  = *(const v4f*)(arow + ka);
    *(v4f*)(fa + 4)  = *(const v4f*)(arow + ka + 4);
    *(v4f*)(fa + 8)  = *(const v4f*)(arow + ka + 8);
    *(v4f*)(fa + 12) = *(const v4f*)(arow + ka + 12);
    v16bf ahi, alo;
#pragma unroll
    for (int e = 0; e < 16; ++e) {
      const __bf16 h = (__bf16)fa[e];
      ahi[e] = h;
      alo[e] = (__bf16)(fa[e] - (float)h);
    }
    // B fragment: 16 contiguous K at fixed N
    const int kb = kk + lh * 16;
    const v16bf bhi = *(const v16bf*)(bhrow + kb);
    const v16bf blo = *(const v16bf*)(blrow + kb);
    acc = __builtin_amdgcn_wmma_f32_16x16x32_bf16(false, ahi, false, bhi, (short)0, acc, false, false);
    acc = __builtin_amdgcn_wmma_f32_16x16x32_bf16(false, ahi, false, blo, (short)0, acc, false, false);
    acc = __builtin_amdgcn_wmma_f32_16x16x32_bf16(false, alo, false, bhi, (short)0, acc, false, false);
  }
  const int n = n0 + lm;
  const float bias = b[n] + b2[n];
#pragma unroll
  for (int v = 0; v < 8; ++v) {
    const int m = m0 + v + lh * 8;     // C/D layout: VGPR v -> M = v + (lanehalf)*8
    out[(size_t)m * HID + n] = acc[v] + bias;
  }
}

// ---------------------------------------------------------------------------
// Persistent recurrence kernel. 16 blocks x 512 threads (16 waves); block owns
// a 64(row) x 64(col) slice of the state; one wave per 16x16 tile.
// Per K-chunk (64), all four operand slabs (A hi/lo, B hi/lo; 8 KB each) are
// staged into LDS with async global->LDS b128 copies, double-buffered:
//   issue(c+1) -> s_wait_asynccnt 4 (chunk c done, c+1 in flight) -> barrier
//   -> 6 WMMAs from LDS -> barrier.
// Monotonic atomic-counter grid barrier between steps.
// ---------------------------------------------------------------------------
__global__ void __launch_bounds__(512, 1) rnn_steps(
    const __bf16* __restrict__ wthi, const __bf16* __restrict__ wtlo,
    __bf16* __restrict__ hhi, __bf16* __restrict__ hlo,
    float* __restrict__ out, float* __restrict__ hlast,
    unsigned* __restrict__ cnt) {
  extern __shared__ __bf16 smem[];     // [2 bufs][4 slabs][64 rows * KCH]
  constexpr int KCH  = 64;             // K elements staged per chunk
  constexpr int NKCH = HID / KCH;      // 16 chunks
  constexpr int SLAB = 64 * KCH;       // 4096 bf16 = 8 KB per slab

  const int tid  = threadIdx.x;
  const int lane = tid & 31;
  const int wave = tid >> 5;           // 0..15
  const int mt = wave & 3;             // batch tile 0..3
  const int nl = wave >> 2;            // local col tile 0..3
  const int m0 = mt * 16;
  const int lm = lane & 15;
  const int lh = lane >> 4;
  const int n0blk = blockIdx.x * 64;
  const int n  = n0blk + nl * 16 + lm;

  // copy-engine mapping: 512 threads x 4 slabs x 16B == one 32 KB chunk
  const int crow = tid >> 3;           // 0..63 (row within slab)
  const int ccol = (tid & 7) * 8;      // bf16 element offset (16B segments)
  const int ldo  = crow * KCH + ccol;  // LDS element offset within slab

  const __bf16* bWh = wthi + (size_t)(n0blk + crow) * HID + ccol;
  const __bf16* bWl = wtlo + (size_t)(n0blk + crow) * HID + ccol;

  for (int t = 0; t < SEQ; ++t) {
    const int rb = (t & 1);            // h read buffer parity
    const int wb = rb ^ 1;             // h write buffer parity
    const __bf16* bAh = hhi + (size_t)rb * BATCH * HID + (size_t)crow * HID + ccol;
    const __bf16* bAl = hlo + (size_t)rb * BATCH * HID + (size_t)crow * HID + ccol;

    auto issue = [&](int c) {
      __bf16* db = smem + (size_t)(c & 1) * 4 * SLAB;
      const int kk = c * KCH;
      async16(bAh + kk, db + 0 * SLAB + ldo);
      async16(bAl + kk, db + 1 * SLAB + ldo);
      async16(bWh + kk, db + 2 * SLAB + ldo);
      async16(bWl + kk, db + 3 * SLAB + ldo);
    };

    issue(0);
    v8f acc = {};
    for (int c = 0; c < NKCH; ++c) {
      if (c + 1 < NKCH) {
        issue(c + 1);
        __builtin_amdgcn_s_wait_asynccnt(4);   // chunk c landed; c+1 in flight
      } else {
        __builtin_amdgcn_s_wait_asynccnt(0);
      }
      __syncthreads();

      const __bf16* cb  = smem + (size_t)(c & 1) * 4 * SLAB;
      const __bf16* sAh = cb + 0 * SLAB + (size_t)(m0 + lm) * KCH;
      const __bf16* sAl = cb + 1 * SLAB + (size_t)(m0 + lm) * KCH;
      const __bf16* sBh = cb + 2 * SLAB + (size_t)(nl * 16 + lm) * KCH;
      const __bf16* sBl = cb + 3 * SLAB + (size_t)(nl * 16 + lm) * KCH;
#pragma unroll
      for (int k2 = 0; k2 < KCH; k2 += 32) {
        const int ka = k2 + lh * 16;
        const v16bf ahi = cat16(*(const v8bf*)(sAh + ka), *(const v8bf*)(sAh + ka + 8));
        const v16bf alo = cat16(*(const v8bf*)(sAl + ka), *(const v8bf*)(sAl + ka + 8));
        const int kb = k2 + lh * 16;
        const v16bf bhi = *(const v16bf*)(sBh + kb);
        const v16bf blo = *(const v16bf*)(sBl + kb);
        acc = __builtin_amdgcn_wmma_f32_16x16x32_bf16(false, ahi, false, bhi, (short)0, acc, false, false);
        acc = __builtin_amdgcn_wmma_f32_16x16x32_bf16(false, ahi, false, blo, (short)0, acc, false, false);
        acc = __builtin_amdgcn_wmma_f32_16x16x32_bf16(false, alo, false, bhi, (short)0, acc, false, false);
      }
      __syncthreads();                 // protect buffer (c&1) before re-fill
    }

    // epilogue: h = tanh(acc + xv); overwrite xv slot with h; stage bf16 hi/lo
    float* orow = out + (size_t)t * BATCH * HID;
    __bf16* hwh = hhi + (size_t)wb * BATCH * HID;
    __bf16* hwl = hlo + (size_t)wb * BATCH * HID;
#pragma unroll
    for (int v = 0; v < 8; ++v) {
      const int m = m0 + v + lh * 8;
      const size_t o = (size_t)m * HID + n;
      const float val = tanhf(acc[v] + orow[o]);
      orow[o] = val;
      const __bf16 h = (__bf16)val;
      hwh[o] = h;
      hwl[o] = (__bf16)(val - (float)h);
      if (t == SEQ - 1) hlast[o] = val;
    }

    // grid-wide barrier (monotonic counter; 16 co-resident workgroups)
    __syncthreads();
    if (threadIdx.x == 0) {
      __threadfence();
      atomicAdd(cnt, 1u);
      const unsigned target = (unsigned)gridDim.x * (unsigned)(t + 1);
      while (__hip_atomic_load(cnt, __ATOMIC_ACQUIRE, __HIP_MEMORY_SCOPE_AGENT) < target) {
        __builtin_amdgcn_s_sleep(1);
      }
    }
    __syncthreads();
  }
}

// ---------------------------------------------------------------------------
extern "C" void kernel_launch(void* const* d_in, const int* in_sizes, int n_in,
                              void* d_out, int out_size, void* d_ws, size_t ws_size,
                              hipStream_t stream) {
  const float* x  = (const float*)d_in[0];   // (512, 64, 256)
  const float* V  = (const float*)d_in[1];   // (256, 1024)
  const float* W  = (const float*)d_in[2];   // (1024, 1024)
  const float* b  = (const float*)d_in[3];   // (1024,)
  const float* b2 = (const float*)d_in[4];   // (1024,)
  float* out = (float*)d_out;                // h_seq (512*64*1024) ++ h_last (64*1024)

  char* ws = (char*)d_ws;
  unsigned* cnt  = (unsigned*)(ws + OFF_CNT);
  __bf16*   hhi  = (__bf16*)(ws + OFF_HHI);
  __bf16*   hlo  = (__bf16*)(ws + OFF_HLO);
  __bf16*   wthi = (__bf16*)(ws + OFF_WTHI);
  __bf16*   wtlo = (__bf16*)(ws + OFF_WTLO);
  __bf16*   vthi = (__bf16*)(ws + OFF_VTHI);
  __bf16*   vtlo = (__bf16*)(ws + OFF_VTLO);

  rnn_prep<<<512, 256, 0, stream>>>(W, V, cnt, wthi, wtlo, vthi, vtlo, hhi, hlo);
  rnn_xv<<<16384, 256, 0, stream>>>(x, vthi, vtlo, b, b2, out);

  const size_t ldsBytes = 2u * 4u * 64u * 64u * sizeof(__bf16);   // 64 KB
  rnn_steps<<<16, 512, ldsBytes, stream>>>(wthi, wtlo, hhi, hlo, out,
                                           out + (size_t)SEQ * BATCH * HID, cnt);
}
